// Flgc2d_16011638079621
// MI455X (gfx1250) — compile-verified, matrix operand order checked by reading, not run on hardware
//
#include <hip/hip_runtime.h>
#include <hip/hip_bf16.h>
#include <cmath>
#include <cstdint>

// ---------------------------------------------------------------------------
// MI455X / gfx1250 implementation notes
//
// Roofline: out = 0.5*G + 0.5*G_new with G 16384x16384 fp32 -> 1 GB read +
// 1 GB write is the floor (~90us @ 23.3 TB/s).  G_new is ~3000x sparse
// (kn=10 ones per edge), so it is computed as a sparse scatter, NOT a dense
// 2.2-TFLOP matmul.  Softmax is monotonic per row -> top-k of the raw map
// equals top-k of softmax(softmax(map)), so softmax is skipped entirely.
//
// CDNA5-specific paths used:
//  * v_wmma_f32_16x16x32_f16 : one WMMA per edge computes the full 16x16
//    outer product w*w^T (w padded kn->16) = all pairwise corrections.
//  * global_load_async_to_lds_b128 + s_wait_asynccnt : async global->LDS
//    staging of each 64 KB row for the top-k passes (ASYNCcnt tracked).
//  * ds_bpermute (via __shfl) to broadcast the kn vertex indices inside the
//    scatter wave instead of re-loading them from global per pair.
//  * global_prefetch_b8 + nontemporal b128 stores in the streaming merge.
// ---------------------------------------------------------------------------

typedef __attribute__((ext_vector_type(16))) _Float16 v16h;
typedef __attribute__((ext_vector_type(8)))  float    v8f;
typedef __attribute__((ext_vector_type(4)))  float    f32x4;  // native vec4

#define TPB 256        // 8 wave32s per block
#define MAX_KN 16      // WMMA 16x16 tile bounds kn; reference uses kn=10

// Low 32 bits of a generic pointer to LDS == byte offset within the wave's
// LDS allocation (SHARED aperture puts the LDS offset in the low DWORD).
__device__ __forceinline__ uint32_t lds_offset(const void* p) {
    return (uint32_t)(uintptr_t)p;
}

// ---------------- K0: zero the per-vertex degree counters -------------------
__global__ void k0_zero_dv(int* __restrict__ DV, int v_n) {
    int i = blockIdx.x * blockDim.x + threadIdx.x;
    if (i < v_n) DV[i] = 0;
}

// ---------------- K1: row-wise top-k (one block per edge/row) ---------------
// Stage the v_n-float row in LDS (64 KB for v_n=16384; WGP has 320 KB), then
// kn argmax passes over LDS.  Global traffic: the map is read exactly once,
// via the CDNA5 async global->LDS DMA path (no VGPR round trip).
__global__ void k1_topk_rows(const float* __restrict__ map,
                             const int* __restrict__ knp,
                             int v_n,
                             int* __restrict__ eIdx,   // [e_n][MAX_KN]
                             int* __restrict__ DV) {
    extern __shared__ char smem_raw[];
    float* row  = (float*)smem_raw;            // v_n floats
    float* redV = row + v_n;                   // TPB floats
    int*   redI = (int*)(redV + TPB);          // TPB ints

    int kn = *knp; if (kn > MAX_KN) kn = MAX_KN;
    const int tid = threadIdx.x;
    const float* src = map + (size_t)blockIdx.x * (size_t)v_n;

    if (((v_n & 3) == 0) && ((((uintptr_t)src) & 15) == 0)) {
        // Async global->LDS: 16B per lane per issue, ASYNCcnt-tracked.
        const uint32_t lbase = lds_offset(row);
        for (int i = tid * 4; i + 3 < v_n; i += TPB * 4) {
            uint32_t laddr = lbase + (uint32_t)i * 4u;
            uint64_t gaddr = (uint64_t)(uintptr_t)(src + i);
            asm volatile("global_load_async_to_lds_b128 %0, %1, off"
                         :: "v"(laddr), "v"(gaddr) : "memory");
        }
        asm volatile("s_wait_asynccnt 0x0" ::: "memory");
    } else {
        for (int i = tid; i < v_n; i += TPB) row[i] = src[i];
    }
    __syncthreads();

    for (int k = 0; k < kn; ++k) {
        float bv = -INFINITY; int bi = 0;
        for (int i = tid; i < v_n; i += TPB) {
            float v = row[i];
            if (v > bv) { bv = v; bi = i; }
        }
        redV[tid] = bv; redI[tid] = bi;
        __syncthreads();
        for (int s = TPB >> 1; s > 0; s >>= 1) {
            if (tid < s && redV[tid + s] > redV[tid]) {
                redV[tid] = redV[tid + s];
                redI[tid] = redI[tid + s];
            }
            __syncthreads();
        }
        if (tid == 0) {
            int w = redI[0];
            eIdx[blockIdx.x * MAX_KN + k] = w;
            row[w] = -INFINITY;               // mask winner for next pass
            atomicAdd(&DV[w], 1);             // vertex degree
        }
        __syncthreads();
    }
}

// ---------------- K2: scaled inverse sqrt degree ----------------------------
// w[v] = sqrt(theta/kn) * DV[v]^-1/2   (theta = 0.5, invDE = 1/kn),
// so a pair contribution is exactly w[u]*w[v].
__global__ void k2_inv_dv(const int* __restrict__ DV,
                          const int* __restrict__ knp,
                          float* __restrict__ invDVs, int v_n) {
    int i = blockIdx.x * blockDim.x + threadIdx.x;
    if (i >= v_n) return;
    int kn = *knp; if (kn > MAX_KN) kn = MAX_KN; if (kn < 1) kn = 1;
    int dv = DV[i];
    invDVs[i] = (dv > 0) ? sqrtf(0.5f / (float)kn) * rsqrtf((float)dv) : 0.0f;
}

// ---------------- K3: out = 0.5*G (streaming, b128, prefetch, NT) -----------
__global__ void k3_merge_scale(const float* __restrict__ G,
                               float* __restrict__ out, size_t n) {
    const size_t stride = (size_t)gridDim.x * blockDim.x;
    const size_t i0 = (size_t)blockIdx.x * blockDim.x + threadIdx.x;
    const size_t n4 = n >> 2;
    const f32x4* g4 = (const f32x4*)G;
    f32x4* o4 = (f32x4*)out;
    for (size_t k = i0; k < n4; k += stride) {
        __builtin_prefetch(g4 + k + stride, 0, 0);   // global_prefetch_b8
        f32x4 v = g4[k] * 0.5f;
        __builtin_nontemporal_store(v, o4 + k);      // stream-once output
    }
    for (size_t k = (n4 << 2) + i0; k < n; k += stride)
        out[k] = 0.5f * G[k];
}

// ---------------- K4: per-edge WMMA outer product + sparse scatter ----------
// One wave32 per edge.  A[:,0] = w, B[0,:] = w (all other K slices zero), so
// D = A*B + 0 = w*w^T: every pairwise correction of this edge from a single
// v_wmma_f32_16x16x32_f16.  f16 A/B layout (ISA 7.12.2): lanes 0-15 hold
// M(=N)=lane with K=0,1 packed in element 0; lanes 16-31 carry K>=8 -> zero.
// D layout: VGPR r, lanes 0-15 -> (M=r, N=lane), lanes 16-31 -> (M=r+8,
// N=lane-16).  Vertex indices are loaded once (lane j holds ei[j]) and
// broadcast with ds_bpermute instead of per-pair global loads.
__global__ void k4_edge_outer_scatter(const int* __restrict__ eIdx,
                                      const float* __restrict__ invDVs,
                                      const int* __restrict__ knp,
                                      float* __restrict__ out,
                                      int e_n, int v_n) {
    int kn = *knp; if (kn > MAX_KN) kn = MAX_KN;
    const int lane = threadIdx.x & 31;
    const int edge = (int)((blockIdx.x * blockDim.x + threadIdx.x) >> 5);
    if (edge >= e_n) return;                 // wave-uniform exit

    const int* ei = eIdx + (size_t)edge * MAX_KN;

    // Lane j (j < kn) holds vertex index j of this edge; w value for lanes
    // 0..kn-1, zero elsewhere (kn <= 16, so lanes 16-31 are zero).
    int   myidx = (lane < kn) ? ei[lane] : 0;
    float wf    = (lane < kn) ? invDVs[myidx] : 0.0f;

    v16h a = {};                             // K=1..15 slices are zero
    a[0] = (_Float16)wf;                     // K=0 slice = w

    v8f c = {};
    v8f d = __builtin_amdgcn_wmma_f32_16x16x32_f16(
        /*neg_a=*/false, a, /*neg_b=*/false, a,
        /*c_mod=*/(short)0, c, /*reuse_a=*/false, /*reuse_b=*/false);

    const int N   = lane & 15;
    const int mhi = (lane >> 4) * 8;
    const int v   = __shfl(myidx, N, 32);    // ds_bpermute broadcast
    #pragma unroll
    for (int r = 0; r < 8; ++r) {
        const int M = mhi + r;
        const int u = __shfl(myidx, M, 32);  // ds_bpermute broadcast
        if (M < kn && N < kn) {
            size_t off = (size_t)u * (size_t)v_n + (size_t)v;
            atomicAdd(out + off, d[r]);      // global_atomic_add_f32
        }
    }
}

// ---------------------------------------------------------------------------
extern "C" void kernel_launch(void* const* d_in, const int* in_sizes, int n_in,
                              void* d_out, int out_size, void* d_ws, size_t ws_size,
                              hipStream_t stream) {
    const float* map = (const float*)d_in[0];   // [e_n, v_n] fp32
    const float* G   = (const float*)d_in[1];   // [v_n, v_n] fp32
    const int*   knp = (const int*)d_in[2];     // scalar kn (device)
    float* out = (float*)d_out;                 // [v_n, v_n] fp32

    const int v_n = (int)(std::sqrt((double)in_sizes[1]) + 0.5);
    const int e_n = in_sizes[0] / v_n;

    // Workspace layout: DV[v_n] ints | invDVs[v_n] floats | eIdx[e_n*MAX_KN]
    int*   DV     = (int*)d_ws;
    float* invDVs = (float*)(DV + v_n);
    int*   eIdx   = (int*)(invDVs + v_n);
    (void)n_in; (void)ws_size; (void)out_size;

    k0_zero_dv<<<(v_n + TPB - 1) / TPB, TPB, 0, stream>>>(DV, v_n);

    const size_t shmem = (size_t)v_n * sizeof(float)
                       + (size_t)TPB * (sizeof(float) + sizeof(int));
    k1_topk_rows<<<e_n, TPB, shmem, stream>>>(map, knp, v_n, eIdx, DV);

    k2_inv_dv<<<(v_n + TPB - 1) / TPB, TPB, 0, stream>>>(DV, knp, invDVs, v_n);

    const size_t n = (size_t)v_n * (size_t)v_n;
    k3_merge_scale<<<4096, TPB, 0, stream>>>(G, out, n);

    const int waves_per_block = TPB / 32;
    const int blocks = (e_n + waves_per_block - 1) / waves_per_block;
    k4_edge_outer_scatter<<<blocks, TPB, 0, stream>>>(eIdx, invDVs, knp, out,
                                                      e_n, v_n);
}
